// CrossTransformer_42949672960510
// MI455X (gfx1250) — matrix operand order repaired and checked by reading, not compiled, not run on hardware
//
#include <hip/hip_runtime.h>
#include <hip/hip_bf16.h>

// ---------------------------------------------------------------------------
// CrossTransformer block for MI455X (gfx1250, wave32, WMMA + TDM).
// GEMMs: v_wmma_f32_16x16x32_f16; A-tiles streamed by the Tensor Data Mover,
// double-buffered in LDS so TDM(i+1) overlaps WMMA compute on tile i.
// ---------------------------------------------------------------------------

typedef _Float16 half_t;
typedef __attribute__((ext_vector_type(16))) _Float16 v16h;
typedef __attribute__((ext_vector_type(8)))  _Float16 v8h;
typedef __attribute__((ext_vector_type(8)))  float    v8f;
typedef unsigned int u32x4 __attribute__((ext_vector_type(4)));
typedef int          i32x8 __attribute__((ext_vector_type(8)));
typedef int          i32x4 __attribute__((ext_vector_type(4)));

#define Bt  4
#define SQl 1024
#define SKl 2048
#define Cc  1024
#define Hh  16
#define DHl 64

// TDM availability (5-arg builtin on ROCm7.2/clang-22; 6-arg on clang-23+)
#if defined(__has_builtin)
#if __has_builtin(__builtin_amdgcn_tensor_load_to_lds) && \
    __has_builtin(__builtin_amdgcn_s_wait_tensorcnt)
#define HAVE_TDM 1
#endif
#endif

// ---- workspace layout (bytes) ----
static const size_t O_W16_QKV   = 0;                           // 3072*1024 f16
static const size_t O_W16_SAOUT = O_W16_QKV   + 6291456;       // 1024*1024 f16
static const size_t O_W16_CAQ   = O_W16_SAOUT + 2097152;
static const size_t O_W16_CAK   = O_W16_CAQ   + 2097152;
static const size_t O_W16_CAV   = O_W16_CAK   + 2097152;
static const size_t O_W16_CAOUT = O_W16_CAV   + 2097152;
static const size_t O_W16_FF1   = O_W16_CAOUT + 2097152;       // 2048*1024 f16
static const size_t O_W16_FF2   = O_W16_FF1   + 4194304;       // 1024*2048 f16
static const size_t O_LN16      = O_W16_FF2   + 4194304;       // B*SQ*2C f16 (LN/SiLU out, also ctx16)
static const size_t O_QKV16     = O_LN16      + 16777216;      // B*SQ*3072 f16 (also reused for q16)
static const size_t O_XSA       = O_QKV16     + 25165824;      // B*SQ*C f32
static const size_t O_XCA       = O_XSA       + 16777216;      // B*SQ*C f32
static const size_t O_O16       = O_XCA       + 16777216;      // B*SQ*C f16 (attn output, reused)
static const size_t O_K16       = O_O16       + 8388608;       // B*SK*C f16
static const size_t O_V16       = O_K16       + 16777216;      // B*SK*C f16
static const size_t O_F1        = O_V16       + 16777216;      // B*SQ*2C f32
static const size_t O_ESA       = O_F1        + 33554432;      // B*H*SQ*SQ f32; reused as p_ca f16

// ---------------------------------------------------------------------------
// f32 -> f16 conversion (weights / contexts)
// ---------------------------------------------------------------------------
__global__ void f32_to_f16_kernel(const float* __restrict__ in,
                                  half_t* __restrict__ out, long n) {
  long i = (long)blockIdx.x * 256 + threadIdx.x;
  long stride = (long)gridDim.x * 256;
  for (; i < n; i += stride) out[i] = (half_t)in[i];
}

// ---------------------------------------------------------------------------
// Row LayerNorm (+ optional SiLU), f32 in -> f16 out.  One block per row.
// ---------------------------------------------------------------------------
__global__ __launch_bounds__(256) void ln_kernel(
    const float* __restrict__ in, const float* __restrict__ g,
    const float* __restrict__ bta, half_t* __restrict__ out,
    int N, int do_silu) {
  const int row = blockIdx.x;
  const int tid = threadIdx.x;
  const int E = N >> 8;
  const float* rp = in + (long)row * N;
  float v[8];
  float s = 0.f, s2 = 0.f;
  for (int i = 0; i < E; ++i) {
    float x = rp[tid + (i << 8)];
    v[i] = x; s += x; s2 += x * x;
  }
  __shared__ float red[256], red2[256];
  red[tid] = s; red2[tid] = s2;
  __syncthreads();
  for (int off = 128; off > 0; off >>= 1) {
    if (tid < off) { red[tid] += red[tid + off]; red2[tid] += red2[tid + off]; }
    __syncthreads();
  }
  float mu  = red[0] / (float)N;
  float var = red2[0] / (float)N - mu * mu;
  float rs  = rsqrtf(var + 1e-5f);
  half_t* op = out + (long)row * N;
  for (int i = 0; i < E; ++i) {
    int j = tid + (i << 8);
    float y = (v[i] - mu) * rs * g[j] + bta[j];
    if (do_silu) y = y / (1.f + __expf(-y));
    op[j] = (half_t)y;
  }
}

// ---------------------------------------------------------------------------
// Fused scale + softmax over a row, f32 in -> f16 out (in-place safe: row is
// register-resident; all reads precede the post-reduction writes).
// ---------------------------------------------------------------------------
__global__ __launch_bounds__(256) void softmax_kernel(
    const float* __restrict__ in, long inStride,
    half_t* __restrict__ out, long outStride,
    int len, float scale) {
  const long row = blockIdx.x;
  const int tid = threadIdx.x;
  const int E = len >> 8;
  const float* rp = in + row * inStride;
  float v[8];
  float m = -3.4e38f;
  for (int i = 0; i < E; ++i) {
    float x = rp[tid + (i << 8)] * scale;
    v[i] = x; m = fmaxf(m, x);
  }
  __shared__ float red[256];
  red[tid] = m; __syncthreads();
  for (int off = 128; off > 0; off >>= 1) {
    if (tid < off) red[tid] = fmaxf(red[tid], red[tid + off]);
    __syncthreads();
  }
  m = red[0];
  __syncthreads();
  float s = 0.f;
  for (int i = 0; i < E; ++i) { v[i] = __expf(v[i] - m); s += v[i]; }
  red[tid] = s; __syncthreads();
  for (int off = 128; off > 0; off >>= 1) {
    if (tid < off) red[tid] += red[tid + off];
    __syncthreads();
  }
  float inv = 1.f / red[0];
  half_t* op = out + row * outStride;
  for (int i = 0; i < E; ++i) op[tid + (i << 8)] = (half_t)(v[i] * inv);
}

// ---------------------------------------------------------------------------
// Batched WMMA GEMM, block tile 128x64x64, 8 waves, 2x2 WMMA tiles per wave,
// two 16x16x32 K-substeps per stage (8 static v_wmma per K iteration).
//   TRANSB == 0 : B stored [n][k]  (Linear weights / K^T)
//   TRANSB == 1 : B stored [k][n]  (attention V)
// Fragment-major LDS (As[row][k], Bs[col][k], stride 72) -> all fragment
// loads are paired ds_load_b128.  Software pipeline: double-buffered LDS;
// wave 0 streams A-tiles with the Tensor Data Mover (pad_interval/pad_amount
// produce the 72-half LDS stride), s_wait_tensorcnt<=1 retires tile i while
// tile i+1 stays in flight under the WMMA compute.
// ---------------------------------------------------------------------------
#define BM 128
#define BN 64
#define BK2 64
#define LDAS 72
#define LDBS 72

#ifdef HAVE_TDM
__device__ __forceinline__ void tdm_load_a_tile(
    const half_t* gptr, unsigned ldsOff, int Ktot, int lda) {
  unsigned long long ga = (unsigned long long)(uintptr_t)gptr;
  u32x4 g0;
  g0[0] = 1u;                                   // count=1, user descriptor
  g0[1] = ldsOff;                               // lds_addr (bytes)
  g0[2] = (unsigned)(ga & 0xffffffffu);         // global_addr[31:0]
  g0[3] = (unsigned)((ga >> 32) & 0x01ffffffu)  // global_addr[56:32]
        | (2u << 30);                           // type = 2 (image)
  i32x8 g1;
  // data_size=1 (2B), pad_enable, pad_interval=32 DWORDs (code 4),
  // pad_amount=4 DWORDs (code 3) -> LDS row stride 64+8 = 72 halves
  g1[0] = (1 << 16) | (1 << 20) | (4 << 22) | (3 << 25);
  g1[1] = (Ktot & 0xffff) << 16;                          // tensor_dim0 lo
  g1[2] = ((Ktot >> 16) & 0xffff) | (BM << 16);           // dim0 hi | tensor_dim1 lo
  g1[3] = (BK2 << 16);                                    // dim1 hi=0 | tile_dim0=64
  g1[4] = BM;                                             // tile_dim1=128, tile_dim2=0
  g1[5] = lda;                                            // tensor_dim0_stride lo
  g1[6] = 0;
  g1[7] = 0;
  i32x4 g2 = {0, 0, 0, 0};
  i32x4 g3 = {0, 0, 0, 0};
#if defined(__clang_major__) && (__clang_major__ >= 23)
  i32x8 g4 = {0, 0, 0, 0, 0, 0, 0, 0};
  __builtin_amdgcn_tensor_load_to_lds(g0, g1, g2, g3, g4, 0);
#else
  __builtin_amdgcn_tensor_load_to_lds(g0, g1, g2, g3, 0);
#endif
}
#endif

template<int TRANSB, int OUT16>
__global__ __launch_bounds__(256) void wmma_gemm_kernel(
    const half_t* __restrict__ A, long sAb, long sAh, int lda,
    const half_t* __restrict__ Bw, long sBb, long sBh, int ldb,
    float* __restrict__ Cf, half_t* __restrict__ Ch, long sCb, long sCh, int ldc,
    const float* __restrict__ bias, const float* __restrict__ resid, int ldres,
    int K, int Hdiv) {
  __shared__ __align__(16) half_t As[2][BM * LDAS];
  __shared__ __align__(16) half_t Bs[2][BN * LDBS];

  const int tid    = threadIdx.x;
  const int lane   = tid & 31;
  const int wave   = tid >> 5;
  const int laneLo = lane & 15;
  const int laneHi = (lane >> 4) & 1;
  const int wm     = (wave & 3) * 32;   // 4 waves along M
  const int wn     = (wave >> 2) * 32;  // 2 waves along N

  const int z  = blockIdx.z;
  const int zb = z / Hdiv;
  const int zh = z % Hdiv;
  const long offA = (long)zb * sAb + (long)zh * sAh;
  const long offB = (long)zb * sBb + (long)zh * sBh;
  const long offC = (long)zb * sCb + (long)zh * sCh;

  const int blockM = blockIdx.y * BM;
  const int blockN = blockIdx.x * BN;

  v8f acc[2][2] = {};

  const half_t* Abase = A + offA + (long)blockM * lda;
  const half_t* Bbase = Bw + offB;

  // ---- staging helpers (lambdas keep both pipeline stages identical) ----
  auto stageB = [&](int kb, int buf) {
    if (TRANSB == 0) {
      const int n  = tid >> 2;
      const int ks = (tid & 3) * 16;
      const half_t* bp = Bbase + (long)(blockN + n) * ldb + kb + ks;
      *(v8h*)&Bs[buf][n * LDBS + ks]     = *(const v8h*)bp;
      *(v8h*)&Bs[buf][n * LDBS + ks + 8] = *(const v8h*)(bp + 8);
      __builtin_prefetch(bp + 2 * BK2, 0, 3);
    } else {
      const int kk = tid >> 2;
      const int ns = (tid & 3) * 16;
      const half_t* bp = Bbase + (long)(kb + kk) * ldb + blockN + ns;
      v8h x0 = *(const v8h*)bp;
      v8h x1 = *(const v8h*)(bp + 8);
#pragma unroll
      for (int i = 0; i < 8; ++i) {
        Bs[buf][(ns + i) * LDBS + kk]     = x0[i];
        Bs[buf][(ns + 8 + i) * LDBS + kk] = x1[i];
      }
    }
  };
  auto stageA = [&](int kb, int buf) {
#ifdef HAVE_TDM
    if (wave == 0)
      tdm_load_a_tile(Abase + kb, (unsigned)(uintptr_t)(&As[buf][0]), K, lda);
#else
    const int r  = tid >> 1;
    const int ks = (tid & 1) * 32;
    const half_t* ap = Abase + (long)r * lda + kb + ks;
#pragma unroll
    for (int i = 0; i < 4; ++i)
      *(v8h*)&As[buf][r * LDAS + ks + i * 8] = *(const v8h*)(ap + i * 8);
#endif
  };

  // ---- prologue: stage tile 0 ----
  stageB(0, 0);
  stageA(0, 0);

  const int nIter = K / BK2;
  for (int it = 0; it < nIter; ++it) {
    const int buf = it & 1;
    __syncthreads();  // retires compute(it-1); publishes B(it)
    if (it + 1 < nIter) {
      stageB((it + 1) * BK2, buf ^ 1);
      stageA((it + 1) * BK2, buf ^ 1);   // TDM A(it+1) stays in flight
#ifdef HAVE_TDM
      if (wave == 0) __builtin_amdgcn_s_wait_tensorcnt(1);  // A(it) done
#endif
    } else {
#ifdef HAVE_TDM
      if (wave == 0) __builtin_amdgcn_s_wait_tensorcnt(0);
#endif
    }
    __syncthreads();  // publishes A(it)

    // ---- 2 K-substeps x (2x2) WMMA on buffer `buf` ----
#pragma unroll
    for (int kk2 = 0; kk2 < BK2; kk2 += 32) {
      v16h afrag[2];
#pragma unroll
      for (int mi = 0; mi < 2; ++mi) {
        const half_t* ap = &As[buf][(wm + mi * 16 + laneLo) * LDAS + kk2 + laneHi * 8];
#pragma unroll
        for (int j = 0; j < 8; ++j) {
          afrag[mi][j]     = ap[j];
          afrag[mi][j + 8] = ap[16 + j];
        }
      }
      v16h bfrag[2];
#pragma unroll
      for (int ni = 0; ni < 2; ++ni) {
        const half_t* bp = &Bs[buf][(wn + ni * 16 + laneLo) * LDBS + kk2 + laneHi * 8];
#pragma unroll
        for (int j = 0; j < 8; ++j) {
          bfrag[ni][j]     = bp[j];
          bfrag[ni][j + 8] = bp[16 + j];
        }
      }
#pragma unroll
      for (int mi = 0; mi < 2; ++mi)
#pragma unroll
        for (int ni = 0; ni < 2; ++ni)
          acc[mi][ni] = __builtin_amdgcn_wmma_f32_16x16x32_f16(
              false, afrag[mi], false, bfrag[ni], (short)0, acc[mi][ni],
              false, false);
    }
  }

  // ---- epilogue: D layout -> row = r + 8*laneHi, col = laneLo ----
#pragma unroll
  for (int mi = 0; mi < 2; ++mi) {
#pragma unroll
    for (int ni = 0; ni < 2; ++ni) {
      const int col = blockN + wn + ni * 16 + laneLo;
#pragma unroll
      for (int r = 0; r < 8; ++r) {
        const int row = blockM + wm + mi * 16 + laneHi * 8 + r;
        float val = acc[mi][ni][r];
        if (bias)  val += bias[col];
        if (resid) val += resid[(long)row * ldres + col];
        if (OUT16) Ch[offC + (long)row * ldc + col] = (half_t)val;
        else       Cf[offC + (long)row * ldc + col] = val;
      }
    }
  }
}

// ---------------------------------------------------------------------------
// host-side launch helpers
// ---------------------------------------------------------------------------
static void launch_gemm(hipStream_t s,
    const half_t* A, long sAb, long sAh, int lda,
    const half_t* B, long sBb, long sBh, int ldb,
    float* Cf, half_t* Ch, long sCb, long sCh, int ldc,
    const float* bias, const float* resid, int ldres,
    int M, int N, int K, int batches, int Hdiv, int transB) {
  dim3 g(N / BN, M / BM, batches);
  if (transB) {
    if (Ch) wmma_gemm_kernel<1, 1><<<g, 256, 0, s>>>(A, sAb, sAh, lda, B, sBb, sBh, ldb,
                Cf, Ch, sCb, sCh, ldc, bias, resid, ldres, K, Hdiv);
    else    wmma_gemm_kernel<1, 0><<<g, 256, 0, s>>>(A, sAb, sAh, lda, B, sBb, sBh, ldb,
                Cf, Ch, sCb, sCh, ldc, bias, resid, ldres, K, Hdiv);
  } else {
    if (Ch) wmma_gemm_kernel<0, 1><<<g, 256, 0, s>>>(A, sAb, sAh, lda, B, sBb, sBh, ldb,
                Cf, Ch, sCb, sCh, ldc, bias, resid, ldres, K, Hdiv);
    else    wmma_gemm_kernel<0, 0><<<g, 256, 0, s>>>(A, sAb, sAh, lda, B, sBb, sBh, ldb,
                Cf, Ch, sCb, sCh, ldc, bias, resid, ldres, K, Hdiv);
  }
}

static inline void launch_cvt(hipStream_t s, const float* in, half_t* out, long n) {
  f32_to_f16_kernel<<<2048, 256, 0, s>>>(in, out, n);
}

extern "C" void kernel_launch(void* const* d_in, const int* in_sizes, int n_in,
                              void* d_out, int out_size, void* d_ws, size_t ws_size,
                              hipStream_t stream) {
  (void)in_sizes; (void)n_in; (void)out_size; (void)ws_size;
  const float* queries  = (const float*)d_in[0];
  const float* contexts = (const float*)d_in[1];
  const float* sa_ln_g  = (const float*)d_in[2];
  const float* sa_ln_b  = (const float*)d_in[3];
  const float* sa_qkv_w = (const float*)d_in[4];
  const float* sa_out_w = (const float*)d_in[5];
  const float* sa_out_b = (const float*)d_in[6];
  const float* ca_ln_g  = (const float*)d_in[7];
  const float* ca_ln_b  = (const float*)d_in[8];
  const float* ca_q_w   = (const float*)d_in[9];
  const float* ca_k_w   = (const float*)d_in[10];
  const float* ca_v_w   = (const float*)d_in[11];
  const float* ca_out_w = (const float*)d_in[12];
  const float* ca_out_b = (const float*)d_in[13];
  const float* ff_ln1_g = (const float*)d_in[14];
  const float* ff_ln1_b = (const float*)d_in[15];
  const float* ff_w1    = (const float*)d_in[16];
  const float* ff_ln2_g = (const float*)d_in[17];
  const float* ff_ln2_b = (const float*)d_in[18];
  const float* ff_w2    = (const float*)d_in[19];

  char* ws = (char*)d_ws;
  half_t* w_qkv   = (half_t*)(ws + O_W16_QKV);
  half_t* w_saout = (half_t*)(ws + O_W16_SAOUT);
  half_t* w_caq   = (half_t*)(ws + O_W16_CAQ);
  half_t* w_cak   = (half_t*)(ws + O_W16_CAK);
  half_t* w_cav   = (half_t*)(ws + O_W16_CAV);
  half_t* w_caout = (half_t*)(ws + O_W16_CAOUT);
  half_t* w_ff1   = (half_t*)(ws + O_W16_FF1);
  half_t* w_ff2   = (half_t*)(ws + O_W16_FF2);
  half_t* ln16    = (half_t*)(ws + O_LN16);
  half_t* qkv16   = (half_t*)(ws + O_QKV16);   // reused as q16 in cross-attn
  float*  xsa     = (float*)(ws + O_XSA);
  float*  xca     = (float*)(ws + O_XCA);
  half_t* o16     = (half_t*)(ws + O_O16);
  half_t* k16     = (half_t*)(ws + O_K16);
  half_t* v16     = (half_t*)(ws + O_V16);
  float*  f1      = (float*)(ws + O_F1);
  float*  e_sa    = (float*)(ws + O_ESA);
  half_t* p_any   = (half_t*)(ws + O_ESA);     // f16 probs overlay

  float*  out_x = (float*)d_out;                              // (B,SQ,C)
  float*  out_e = (float*)d_out + (size_t)Bt * SQl * Cc;      // (B,H,SQ,SK)

  const float scale = 0.03125f;  // 1/sqrt(C)
  const long nRowsQ = (long)Bt * SQl;         // 4096
  const long nRowsK = (long)Bt * SKl;         // 8192

  // --- weights fp32 -> fp16 ---
  launch_cvt(stream, sa_qkv_w, w_qkv,   (long)3072 * 1024);
  launch_cvt(stream, sa_out_w, w_saout, (long)1024 * 1024);
  launch_cvt(stream, ca_q_w,   w_caq,   (long)1024 * 1024);
  launch_cvt(stream, ca_k_w,   w_cak,   (long)1024 * 1024);
  launch_cvt(stream, ca_v_w,   w_cav,   (long)1024 * 1024);
  launch_cvt(stream, ca_out_w, w_caout, (long)1024 * 1024);
  launch_cvt(stream, ff_w1,    w_ff1,   (long)2048 * 1024);
  launch_cvt(stream, ff_w2,    w_ff2,   (long)1024 * 2048);

  // ================= Self-attention =================
  ln_kernel<<<(int)nRowsQ, 256, 0, stream>>>(queries, sa_ln_g, sa_ln_b, ln16, Cc, 0);
  launch_gemm(stream, ln16, 0, 0, Cc, w_qkv, 0, 0, Cc,
              nullptr, qkv16, 0, 0, 3072, nullptr, nullptr, 0,
              (int)nRowsQ, 3072, Cc, 1, 1, 0);
  launch_gemm(stream,
              qkv16,      (long)SQl * 3072, 192, 3072,
              qkv16 + 64, (long)SQl * 3072, 192, 3072,
              e_sa, nullptr, (long)Hh * SQl * SQl, (long)SQl * SQl, SQl,
              nullptr, nullptr, 0, SQl, SQl, DHl, Bt * Hh, Hh, 0);
  softmax_kernel<<<Bt * Hh * SQl, 256, 0, stream>>>(e_sa, SQl, p_any, 2 * SQl, SQl, scale);
  launch_gemm(stream,
              p_any, (long)Hh * SQl * 2 * SQl, (long)SQl * 2 * SQl, 2 * SQl,
              qkv16 + 128, (long)SQl * 3072, 192, 3072,
              nullptr, o16, (long)SQl * Cc, DHl, Cc,
              nullptr, nullptr, 0, SQl, DHl, SQl, Bt * Hh, Hh, 1);
  launch_gemm(stream, o16, 0, 0, Cc, w_saout, 0, 0, Cc,
              xsa, nullptr, 0, 0, Cc, sa_out_b, queries, Cc,
              (int)nRowsQ, Cc, Cc, 1, 1, 0);

  // ================= Cross-attention =================
  ln_kernel<<<(int)nRowsQ, 256, 0, stream>>>(xsa, ca_ln_g, ca_ln_b, ln16, Cc, 0);
  launch_gemm(stream, ln16, 0, 0, Cc, w_caq, 0, 0, Cc,
              nullptr, qkv16, 0, 0, Cc, nullptr, nullptr, 0,
              (int)nRowsQ, Cc, Cc, 1, 1, 0);
  launch_cvt(stream, contexts, ln16, (long)Bt * SKl * Cc);
  launch_gemm(stream, ln16, 0, 0, Cc, w_cak, 0, 0, Cc,
              nullptr, k16, 0, 0, Cc, nullptr, nullptr, 0,
              (int)nRowsK, Cc, Cc, 1, 1, 0);
  launch_gemm(stream, ln16, 0, 0, Cc, w_cav, 0, 0, Cc,
              nullptr, v16, 0, 0, Cc, nullptr, nullptr, 0,
              (int)nRowsK, Cc, Cc, 1, 1, 0);
  launch_gemm(stream,
              qkv16, (long)SQl * Cc, DHl, Cc,
              k16,   (long)SKl * Cc, DHl, Cc,
              out_e, nullptr, (long)Hh * SQl * SKl, (long)SQl * SKl, SKl,
              nullptr, nullptr, 0, SQl, SKl, DHl, Bt * Hh, Hh, 0);
  softmax_kernel<<<Bt * Hh * SQl, 256, 0, stream>>>(out_e, SKl, p_any, SKl, SKl, scale);
  launch_gemm(stream,
              p_any, (long)Hh * SQl * SKl, (long)SQl * SKl, SKl,
              v16, (long)SKl * Cc, DHl, Cc,
              nullptr, o16, (long)SQl * Cc, DHl, Cc,
              nullptr, nullptr, 0, SQl, DHl, SKl, Bt * Hh, Hh, 1);
  launch_gemm(stream, o16, 0, 0, Cc, w_caout, 0, 0, Cc,
              xca, nullptr, 0, 0, Cc, ca_out_b, xsa, Cc,
              (int)nRowsQ, Cc, Cc, 1, 1, 0);

  // ================= FFN =================
  ln_kernel<<<(int)nRowsQ, 256, 0, stream>>>(xca, ff_ln1_g, ff_ln1_b, ln16, Cc, 1);
  launch_gemm(stream, ln16, 0, 0, Cc, w_ff1, 0, 0, Cc,
              f1, nullptr, 0, 0, 2 * Cc, nullptr, nullptr, 0,
              (int)nRowsQ, 2 * Cc, Cc, 1, 1, 0);
  ln_kernel<<<(int)nRowsQ, 256, 0, stream>>>(f1, ff_ln2_g, ff_ln2_b, ln16, 2 * Cc, 1);
  launch_gemm(stream, ln16, 0, 0, 2 * Cc, w_ff2, 0, 0, 2 * Cc,
              out_x, nullptr, 0, 0, Cc, nullptr, xca, Cc,
              (int)nRowsQ, Cc, 2 * Cc, 1, 1, 0);
}
